// ImageReconstruction_37005438222627
// MI455X (gfx1250) — matrix operand order repaired and checked by reading, not compiled, not run on hardware
//
#include <hip/hip_runtime.h>

// Problem constants (match reference)
#define BB   64
#define HH   512
#define WW   512
#define KK   256          // codebook size
#define LL   64           // block_len (8x8)
#define WTt  64           // tiles per row (W/8)
#define TT   4096         // tiles per image
#define NN   (BB * TT)    // 262144 patches
#define BETA 35.0f

#define WAVES 8
#define PATCH_PER_BLOCK (WAVES * 16)       // 128
#define NBLOCKS (NN / PATCH_PER_BLOCK)     // 2048

typedef __attribute__((ext_vector_type(16))) __bf16 v16bf;
typedef __attribute__((ext_vector_type(8)))  __bf16 v8bf;
typedef __attribute__((ext_vector_type(8)))  float  v8f;

__global__ __launch_bounds__(256)
void vq_recon_kernel(const float* __restrict__ img,
                     const float* __restrict__ blocks,
                     float* __restrict__ out,
                     float* __restrict__ partial)
{
    __shared__ __align__(16) __bf16 s_codes [KK][LL];   // codes[k][l]   32 KB
    __shared__ __align__(16) __bf16 s_codesT[LL][KK];   // codes[l][k]   32 KB
    __shared__ float  s_b2[KK];                         //                1 KB
    __shared__ __align__(16) __bf16 s_w[WAVES][16][KK]; // weights stage 64 KB
    __shared__ float  s_red[256];

    const int tid = threadIdx.x;

    // ---- Stage codebook into LDS (row-major + transposed) and compute b2 ----
    {
        const float* row = blocks + tid * LL;
        float b2 = 0.f;
        #pragma unroll
        for (int l = 0; l < LL; l += 4) {
            float4 v = *(const float4*)(row + l);
            b2 += v.x*v.x + v.y*v.y + v.z*v.z + v.w*v.w;
            s_codes[tid][l+0] = (__bf16)v.x;  s_codesT[l+0][tid] = (__bf16)v.x;
            s_codes[tid][l+1] = (__bf16)v.y;  s_codesT[l+1][tid] = (__bf16)v.y;
            s_codes[tid][l+2] = (__bf16)v.z;  s_codesT[l+2][tid] = (__bf16)v.z;
            s_codes[tid][l+3] = (__bf16)v.w;  s_codesT[l+3][tid] = (__bf16)v.w;
        }
        s_b2[tid] = b2;
    }
    __syncthreads();

    const int wave = tid >> 5;
    const int lane = tid & 31;
    const int lmod = lane & 15;     // N index / A-row within tile
    const int lhi  = lane >> 4;     // half-wave selector
    const int m_base = blockIdx.x * PATCH_PER_BLOCK + wave * 16;

    // ---- Load A fragments: 16 patches x 64 elems (two K=32 chunks) ----
    // A layout: lane M = lane%16; half j -> K = (lane/16)*16 + j
    // => per chunk c: row kh0 = c*4 + 2*lhi (halves 0..7), row kh0+1 (halves 8..15)
    const int n0  = m_base + lmod;
    const int b0  = n0 >> 12;
    const int t0  = n0 & 4095;
    const int th0 = t0 >> 6;
    const int tw0 = t0 & 63;
    const float* pbase = img + (size_t)b0 * (HH * WW) + th0 * 8 * WW + tw0 * 8;

    v16bf afr[2];
    #pragma unroll
    for (int c = 0; c < 2; ++c) {
        const int kh0 = c * 4 + lhi * 2;
        float4 x0 = *(const float4*)(pbase + kh0 * WW);
        float4 x1 = *(const float4*)(pbase + kh0 * WW + 4);
        float4 y0 = *(const float4*)(pbase + (kh0 + 1) * WW);
        float4 y1 = *(const float4*)(pbase + (kh0 + 1) * WW + 4);
        afr[c][0]  = (__bf16)x0.x; afr[c][1]  = (__bf16)x0.y;
        afr[c][2]  = (__bf16)x0.z; afr[c][3]  = (__bf16)x0.w;
        afr[c][4]  = (__bf16)x1.x; afr[c][5]  = (__bf16)x1.y;
        afr[c][6]  = (__bf16)x1.z; afr[c][7]  = (__bf16)x1.w;
        afr[c][8]  = (__bf16)y0.x; afr[c][9]  = (__bf16)y0.y;
        afr[c][10] = (__bf16)y0.z; afr[c][11] = (__bf16)y0.w;
        afr[c][12] = (__bf16)y1.x; afr[c][13] = (__bf16)y1.y;
        afr[c][14] = (__bf16)y1.z; afr[c][15] = (__bf16)y1.w;
    }

    // ---- GEMM1: ab = patches @ codes^T, keep all 16 K-tiles in registers ----
    v8f sc[16];
    #pragma unroll
    for (int kb = 0; kb < 16; ++kb) {
        v8f acc = {};
        #pragma unroll
        for (int c = 0; c < 2; ++c) {
            // B layout: lane N = lane%16; half j -> K = (lane/16)*16 + j
            const v16bf bfr =
                *(const v16bf*)&s_codes[kb * 16 + lmod][c * 32 + lhi * 16];
            acc = __builtin_amdgcn_wmma_f32_16x16x32_bf16(
                false, afr[c], false, bfr, (short)0, acc, false, false);
        }
        sc[kb] = acc;
    }

    // ---- scores = 2*beta*ab - beta*b2  (a2 cancels in softmax) ----
    #pragma unroll
    for (int kb = 0; kb < 16; ++kb) {
        const float b2v = s_b2[kb * 16 + lmod];
        #pragma unroll
        for (int v = 0; v < 8; ++v)
            sc[kb][v] = 2.0f * BETA * sc[kb][v] - BETA * b2v;
    }

    // ---- softmax over K: per row M = lhi*8 + v; row lives across the 16
    //      lanes sharing lhi, 16 tiles deep. Reduce locally then shfl_xor. ----
    float rmax[8], rinv[8];
    #pragma unroll
    for (int v = 0; v < 8; ++v) {
        float m = sc[0][v];
        #pragma unroll
        for (int kb = 1; kb < 16; ++kb) m = fmaxf(m, sc[kb][v]);
        m = fmaxf(m, __shfl_xor(m, 1, 32));
        m = fmaxf(m, __shfl_xor(m, 2, 32));
        m = fmaxf(m, __shfl_xor(m, 4, 32));
        m = fmaxf(m, __shfl_xor(m, 8, 32));
        rmax[v] = m;
    }
    #pragma unroll
    for (int v = 0; v < 8; ++v) {
        float s = 0.f;
        #pragma unroll
        for (int kb = 0; kb < 16; ++kb) {
            const float e = __expf(sc[kb][v] - rmax[v]);
            sc[kb][v] = e;
            s += e;
        }
        s += __shfl_xor(s, 1, 32);
        s += __shfl_xor(s, 2, 32);
        s += __shfl_xor(s, 4, 32);
        s += __shfl_xor(s, 8, 32);
        rinv[v] = 1.0f / s;
    }

    // ---- weights -> LDS (A-layout staging) + choice penalty ----
    float pen = 0.f;
    #pragma unroll
    for (int kb = 0; kb < 16; ++kb) {
        #pragma unroll
        for (int v = 0; v < 8; ++v) {
            const float w = sc[kb][v] * rinv[v];
            pen += fminf(w, 1.0f - w);
            s_w[wave][lhi * 8 + v][kb * 16 + lmod] = (__bf16)w;
        }
    }
    // same-wave LDS writes/reads are kept in order by hardware (DScnt);
    // other waves use disjoint s_w slabs, so no barrier needed here.

    // ---- GEMM2: out = W(16x256) @ codes(256x64) ----
    v8f o[4] = {};
    #pragma unroll
    for (int c = 0; c < 8; ++c) {
        // A fragment from staged weights: halves 0..7 at [M][c*32 + lhi*16],
        // halves 8..15 at [M][c*32 + lhi*16 + 8]
        const v8bf lo = *(const v8bf*)&s_w[wave][lmod][c * 32 + lhi * 16];
        const v8bf hi = *(const v8bf*)&s_w[wave][lmod][c * 32 + lhi * 16 + 8];
        v16bf af;
        #pragma unroll
        for (int j = 0; j < 8; ++j) { af[j] = lo[j]; af[j + 8] = hi[j]; }
        #pragma unroll
        for (int nt = 0; nt < 4; ++nt) {
            const v16bf bfr =
                *(const v16bf*)&s_codesT[nt * 16 + lmod][c * 32 + lhi * 16];
            o[nt] = __builtin_amdgcn_wmma_f32_16x16x32_bf16(
                false, af, false, bfr, (short)0, o[nt], false, false);
        }
    }

    // ---- fold: scatter reconstructed patches back into the image ----
    #pragma unroll
    for (int v = 0; v < 8; ++v) {
        const int M  = lhi * 8 + v;
        const int np = m_base + M;
        const int bb = np >> 12;
        const int tt = np & 4095;
        const int th = tt >> 6;
        const int tw = tt & 63;
        float* obase = out + (size_t)bb * (HH * WW) + th * 8 * WW + tw * 8;
        #pragma unroll
        for (int nt = 0; nt < 4; ++nt) {
            const int li = nt * 16 + lmod;   // element index in 8x8 patch
            obase[(li >> 3) * WW + (li & 7)] = o[nt][v];
        }
    }

    // ---- deterministic per-block penalty reduction ----
    s_red[tid] = pen;
    __syncthreads();
    #pragma unroll
    for (int s = 128; s > 0; s >>= 1) {
        if (tid < s) s_red[tid] += s_red[tid + s];
        __syncthreads();
    }
    if (tid == 0) partial[blockIdx.x] = s_red[0];
}

__global__ __launch_bounds__(256)
void penalty_reduce_kernel(const float* __restrict__ partial,
                           float* __restrict__ out_scalar)
{
    __shared__ float s[256];
    float a = 0.f;
    for (int i = threadIdx.x; i < NBLOCKS; i += 256) a += partial[i];
    s[threadIdx.x] = a;
    __syncthreads();
    #pragma unroll
    for (int st = 128; st > 0; st >>= 1) {
        if (threadIdx.x < st) s[threadIdx.x] += s[threadIdx.x + st];
        __syncthreads();
    }
    if (threadIdx.x == 0) *out_scalar = s[0] * (1.0f / (float)NN);
}

extern "C" void kernel_launch(void* const* d_in, const int* in_sizes, int n_in,
                              void* d_out, int out_size, void* d_ws, size_t ws_size,
                              hipStream_t stream) {
    const float* img    = (const float*)d_in[0];  // (B,1,H,W) fp32
    const float* blocks = (const float*)d_in[1];  // (K,8,8)  fp32
    float* out     = (float*)d_out;               // B*H*W floats + 1 scalar
    float* partial = (float*)d_ws;                // NBLOCKS floats scratch

    vq_recon_kernel<<<NBLOCKS, 256, 0, stream>>>(img, blocks, out, partial);
    penalty_reduce_kernel<<<1, 256, 0, stream>>>(
        partial, out + (size_t)BB * HH * WW);
}
